// ana_block_36988258353291
// MI455X (gfx1250) — matrix operand order repaired and checked by reading, not compiled, not run on hardware
//
#include <hip/hip_runtime.h>
#include <hip/hip_bf16.h>
#include <math.h>

// ---------------------------------------------------------------------------
// Problem constants (from reference): B=2, C=128, N=4096, H=4, D=32
// ---------------------------------------------------------------------------
#define BB   2
#define CCH  128
#define NPT  4096
#define NH   4
#define HD   32

typedef __attribute__((ext_vector_type(16))) __bf16 v16bf;
typedef __attribute__((ext_vector_type(8)))  __bf16 v8bf;
typedef __attribute__((ext_vector_type(8)))  float  v8f;
typedef __attribute__((ext_vector_type(4)))  int    v4i;

#define BN_RS 0.9999950000374997f      /* 1/sqrt(1+eps), BN with running var 1 */
#define LN_EPS 1e-5f
#define ATT_SCALE 0.17677669529663687f /* 1/sqrt(32) */

// ---- CDNA5 async global->LDS copy (guarded; falls back to sync copy) ------
#if defined(__has_builtin)
#if __has_builtin(__builtin_amdgcn_global_load_async_to_lds_b128) && \
    __has_builtin(__builtin_amdgcn_s_wait_asynccnt)
#define HAVE_ASYNC_LDS 1
#endif
#endif
#ifndef HAVE_ASYNC_LDS
#define HAVE_ASYNC_LDS 0
#endif

#if HAVE_ASYNC_LDS
typedef __attribute__((address_space(1))) v4i gas_v4i;   // global AS pointer
typedef __attribute__((address_space(3))) v4i las_v4i;   // LDS AS pointer
#endif

static __device__ __forceinline__ void cp16_g2s(const __bf16* g, __bf16* s) {
#if HAVE_ASYNC_LDS
  __builtin_amdgcn_global_load_async_to_lds_b128((gas_v4i*)g, (las_v4i*)s, 0, 0);
#else
  *(v8bf*)s = *(const v8bf*)g;
#endif
}
static __device__ __forceinline__ void stage_commit_barrier() {
#if HAVE_ASYNC_LDS
  __builtin_amdgcn_s_wait_asynccnt(0);
#endif
  __syncthreads();
}

static __device__ __forceinline__ v8f wmma_bf16(v16bf a, v16bf b, v8f c) {
  // D = A(16x32) * B(32x16) + C, fp32 accumulate
  return __builtin_amdgcn_wmma_f32_16x16x32_bf16(false, a, false, b, (short)0, c,
                                                 false, false);
}

// Build 16x32 bf16 A-fragment from bf16 row-major [O x ld].
// k order used here (contiguous): lane<16 (row o0+l) holds K = {c0+0..15};
//             lane>=16 (row o0+l-16) holds K = {c0+16..31}.
static __device__ __forceinline__ v16bf load_a_bf(const __bf16* __restrict__ W,
                                                  int ld, int o0, int c0, int lane) {
  const int hi = lane >> 4, r = lane & 15;
  const __bf16* p = W + (size_t)(o0 + r) * ld + c0 + hi * 16;
  v16bf a{};
#pragma unroll
  for (int i = 0; i < 8; ++i) { a[i] = p[i]; a[8 + i] = p[8 + i]; }
  return a;
}

// ---------------------------------------------------------------------------
// fp32 -> bf16 elementwise (weight pre-conversion, runs once per launch)
// ---------------------------------------------------------------------------
__global__ void k_cvt_bf16(const float* __restrict__ src, __bf16* __restrict__ dst, int n) {
  int i = blockIdx.x * blockDim.x + threadIdx.x;
  if (i < n) dst[i] = (__bf16)src[i];
}

// x [B,C,N] fp32 -> xT [B,N,C] bf16 (c contiguous: B-fragment friendly)
__global__ void k_transpose_x(const float* __restrict__ x, __bf16* __restrict__ xT) {
  int i = blockIdx.x * blockDim.x + threadIdx.x;      // B*N*C threads
  int c = i & (CCH - 1);
  int n = (i >> 7) & (NPT - 1);
  int b = i >> 19;
  xT[((size_t)b * NPT + n) * CCH + c] = (__bf16)x[((size_t)b * CCH + c) * NPT + n];
}

// ---------------------------------------------------------------------------
// Q/K projection: out[o,n] = W[o,:]@x[:,n] + bias[o]; scatter to per-head
// layout head[b][h][n][d] (d contiguous), channel o = d*H + h.
// ---------------------------------------------------------------------------
__global__ void k_gemm_qk(const __bf16* __restrict__ Wb, const float* __restrict__ bias,
                          const __bf16* __restrict__ xT, __bf16* __restrict__ outHead) {
  const int lane = threadIdx.x & 31;
  const int hi = lane >> 4, r16 = lane & 15;
  int wid = blockIdx.x * (blockDim.x >> 5) + (threadIdx.x >> 5);  // 0..4095
  int nt = wid & 255, ot = (wid >> 8) & 7, b = wid >> 11;
  int n0 = nt * 16, o0 = ot * 16;

  v8f acc{};
#pragma unroll
  for (int c0 = 0; c0 < CCH; c0 += 32) {
    v16bf a = load_a_bf(Wb, CCH, o0, c0, lane);
    v16bf bf = *(const v16bf*)(xT + ((size_t)b * NPT + n0 + r16) * CCH + c0 + hi * 16);
    acc = wmma_bf16(a, bf, acc);
  }
  int n = n0 + r16;
#pragma unroll
  for (int r = 0; r < 8; ++r) {
    int o = o0 + hi * 8 + r;
    float v = acc[r] + bias[o];
    int d = o >> 2, h = o & 3;
    outHead[(((size_t)b * NH + h) * NPT + n) * HD + d] = (__bf16)v;
  }
}

// Kbf [b][h][m][d] -> Vt [b][h][d][m] (m contiguous, for PV B-fragments)
__global__ void k_transpose_v(const __bf16* __restrict__ Kbf, __bf16* __restrict__ Vt) {
  int i = blockIdx.x * blockDim.x + threadIdx.x;      // B*H*N*D threads
  int d = i & (HD - 1);
  int m = (i >> 5) & (NPT - 1);
  int bh = i >> 17;
  Vt[((size_t)bh * HD + d) * NPT + m] = Kbf[((size_t)bh * NPT + m) * HD + d];
}

// ---------------------------------------------------------------------------
// Attention pass 1: per-row (max, sum-exp) over all 4096 keys.
// Block = 4 waves sharing one (b,h); K staged per 64-key chunk in LDS via
// async copies; each wave owns 16 query rows (one 16x16x32 WMMA per 16 keys).
// D-fragment: lane holds column c=lane%16, rows (lane<16 ? 0..7 : 8..15).
// ---------------------------------------------------------------------------
__global__ void k_attn_lse(const __bf16* __restrict__ Qbf, const __bf16* __restrict__ Kbf,
                           float* __restrict__ rowmax, float* __restrict__ rowsum) {
  __shared__ __align__(64) __bf16 Kls[64 * HD];        // 4 KB
  const int tid = threadIdx.x;
  const int lane = tid & 31, wv = tid >> 5;
  const int hi = lane >> 4, r16 = lane & 15;
  int bh = blockIdx.x >> 6;                            // 512 blocks = 8 bh * 64
  int n0 = (blockIdx.x & 63) * 64 + wv * 16;
  const __bf16* Qh = Qbf + (size_t)bh * NPT * HD;
  const __bf16* Kh = Kbf + (size_t)bh * NPT * HD;

  v16bf aq{};
  {
    const __bf16* p = Qh + (size_t)(n0 + r16) * HD + hi * 16;
#pragma unroll
    for (int i = 0; i < 8; ++i) { aq[i] = p[i]; aq[8 + i] = p[8 + i]; }
  }
  float m[8], s[8];
#pragma unroll
  for (int r = 0; r < 8; ++r) { m[r] = -1e30f; s[r] = 0.f; }

  for (int mc = 0; mc < NPT / 64; ++mc) {
    int m0 = mc * 64;
    // stage K[m0..m0+63][0..31]: 256 x 16B units, 2 per thread
#pragma unroll
    for (int u = 0; u < 2; ++u) {
      int unit = tid * 2 + u;
      int row = unit >> 2, off = (unit & 3) * 8;
      const __bf16* g = Kh + (size_t)(m0 + row) * HD + off;
      cp16_g2s(g, Kls + row * HD + off);
      if (mc + 1 < NPT / 64) __builtin_prefetch((const void*)(g + 64 * HD), 0, 1);
    }
    stage_commit_barrier();
#pragma unroll
    for (int t = 0; t < 4; ++t) {
      v16bf bk = *(const v16bf*)(Kls + (t * 16 + r16) * HD + hi * 16);
      v8f c{};
      c = wmma_bf16(aq, bk, c);
#pragma unroll
      for (int r = 0; r < 8; ++r) {
        float sv = c[r] * ATT_SCALE;
        float nm = fmaxf(m[r], sv);
        s[r] = s[r] * __expf(m[r] - nm) + __expf(sv - nm);
        m[r] = nm;
      }
    }
    __syncthreads();
  }
  // merge the 16 lanes sharing each row (columns are striped across lanes)
#pragma unroll
  for (int off = 1; off < 16; off <<= 1) {
#pragma unroll
    for (int r = 0; r < 8; ++r) {
      float mo = __shfl_xor(m[r], off, 16);
      float so = __shfl_xor(s[r], off, 16);
      float nm = fmaxf(m[r], mo);
      s[r] = s[r] * __expf(m[r] - nm) + so * __expf(mo - nm);
      m[r] = nm;
    }
  }
  if (r16 == 0) {
    int base = bh * NPT + n0 + hi * 8;
#pragma unroll
    for (int r = 0; r < 8; ++r) { rowmax[base + r] = m[r]; rowsum[base + r] = s[r]; }
  }
}

__global__ void k_zero(float* __restrict__ p, int n) {
  int i = blockIdx.x * blockDim.x + threadIdx.x;
  if (i < n) p[i] = 0.f;
}

// ---------------------------------------------------------------------------
// Attention pass 2: recompute S, P = exp(S-max)/sum,
//   colsum[m] += sum_n P[n,m]  (fp32 atomics, one per column per chunk)
//   O = P @ V  (P restaged via per-wave LDS pane into A-fragment layout)
// K and V chunks staged per block in LDS (async). att stored bf16 [b][n][c],
// c = d*4 + h.
// ---------------------------------------------------------------------------
__global__ void k_attn_pv(const __bf16* __restrict__ Qbf, const __bf16* __restrict__ Kbf,
                          const __bf16* __restrict__ Vt,
                          const float* __restrict__ rowmax, const float* __restrict__ rowsum,
                          float* __restrict__ colsum, __bf16* __restrict__ attbf) {
  __shared__ __align__(64) __bf16 Kls[64 * HD];        // 4 KB
  __shared__ __align__(64) __bf16 Vls[HD * 64];        // 4 KB [d][m]
  __shared__ __align__(64) __bf16 Pls[4][16 * 64];     // 8 KB per-wave panes
  const int tid = threadIdx.x;
  const int lane = tid & 31, wv = tid >> 5;
  const int hi = lane >> 4, r16 = lane & 15;
  int bh = blockIdx.x >> 6;
  int b = bh >> 2, h = bh & 3;
  int n0 = (blockIdx.x & 63) * 64 + wv * 16;
  const __bf16* Qh = Qbf + (size_t)bh * NPT * HD;
  const __bf16* Kh = Kbf + (size_t)bh * NPT * HD;
  const __bf16* Vh = Vt + (size_t)bh * HD * NPT;

  v16bf aq{};
  {
    const __bf16* p = Qh + (size_t)(n0 + r16) * HD + hi * 16;
#pragma unroll
    for (int i = 0; i < 8; ++i) { aq[i] = p[i]; aq[8 + i] = p[8 + i]; }
  }
  float rm[8], irs[8];
  {
    int base = bh * NPT + n0 + hi * 8;
#pragma unroll
    for (int r = 0; r < 8; ++r) { rm[r] = rowmax[base + r]; irs[r] = 1.f / rowsum[base + r]; }
  }

  v8f o0f{}, o1f{};
  for (int mc = 0; mc < NPT / 64; ++mc) {
    int m0 = mc * 64;
    // stage K chunk (256 units) and V chunk (256 units), 2+2 per thread
#pragma unroll
    for (int u = 0; u < 2; ++u) {
      int unit = tid * 2 + u;
      int krow = unit >> 2, koff = (unit & 3) * 8;
      const __bf16* gk = Kh + (size_t)(m0 + krow) * HD + koff;
      cp16_g2s(gk, Kls + krow * HD + koff);
      int vrow = unit >> 3, voff = (unit & 7) * 8;
      const __bf16* gv = Vh + (size_t)vrow * NPT + m0 + voff;
      cp16_g2s(gv, Vls + vrow * 64 + voff);
      if (mc + 1 < NPT / 64) {
        __builtin_prefetch((const void*)(gk + 64 * HD), 0, 1);
        __builtin_prefetch((const void*)(gv + 64), 0, 1);
      }
    }
    stage_commit_barrier();

    float colacc[4];
#pragma unroll
    for (int t = 0; t < 4; ++t) {
      v16bf bk = *(const v16bf*)(Kls + (t * 16 + r16) * HD + hi * 16);
      v8f c{};
      c = wmma_bf16(aq, bk, c);
      float ca = 0.f;
#pragma unroll
      for (int r = 0; r < 8; ++r) {
        float pval = __expf(c[r] * ATT_SCALE - rm[r]) * irs[r];
        ca += pval;
        Pls[wv][(hi * 8 + r) * 64 + t * 16 + r16] = (__bf16)pval;  // [row][k]
      }
      colacc[t] = ca;
    }
#pragma unroll
    for (int t = 0; t < 4; ++t) colacc[t] += __shfl_xor(colacc[t], 16, 32);
    if (lane < 16) {
#pragma unroll
      for (int t = 0; t < 4; ++t)
        atomicAdd(&colsum[(size_t)bh * NPT + m0 + t * 16 + lane], colacc[t]);
    }
    // O += P(16x64) @ V(64x32): two K-steps of 32, two d-halves
    // (same-wave LDS ops stay in order: P write -> A-frag read is safe)
#pragma unroll
    for (int sstep = 0; sstep < 2; ++sstep) {
      v16bf ap{};
      const __bf16* pp = &Pls[wv][r16 * 64 + sstep * 32 + hi * 16];
#pragma unroll
      for (int i = 0; i < 8; ++i) { ap[i] = pp[i]; ap[8 + i] = pp[8 + i]; }
      v16bf bv0 = *(const v16bf*)(Vls + (size_t)r16 * 64 + sstep * 32 + hi * 16);
      v16bf bv1 = *(const v16bf*)(Vls + (size_t)(16 + r16) * 64 + sstep * 32 + hi * 16);
      o0f = wmma_bf16(ap, bv0, o0f);
      o1f = wmma_bf16(ap, bv1, o1f);
    }
    __syncthreads();
  }
  // store att[b][n][c], c = d*4 + h
#pragma unroll
  for (int t = 0; t < 2; ++t) {
    v8f of = t ? o1f : o0f;
    int d = t * 16 + r16;
#pragma unroll
    for (int r = 0; r < 8; ++r) {
      int n = n0 + hi * 8 + r;
      attbf[((size_t)b * NPT + n) * CCH + d * 4 + h] = (__bf16)of[r];
    }
  }
}

// ---------------------------------------------------------------------------
// Generic 128-row GEMM over the point axis:
//   out[o,n] = Wb[o,:kdim] @ inB[n,:kdim] + bias[o]   (+ optional BN/ReLU)
// inB bf16 [b][N][ldin] (c contiguous). Output bf16 [b][N][ldout] at channel
// offset ocoff, or fp32 same layout.
// ---------------------------------------------------------------------------
__global__ void k_gemm_row(const __bf16* __restrict__ Wb, const float* __restrict__ bias,
                           const float* __restrict__ bng, const float* __restrict__ bnb,
                           const __bf16* __restrict__ inB, int ldin, int kdim,
                           __bf16* __restrict__ outbf, float* __restrict__ outf,
                           int ldout, int ocoff, int relu) {
  const int lane = threadIdx.x & 31;
  const int hi = lane >> 4, r16 = lane & 15;
  int wid = blockIdx.x * (blockDim.x >> 5) + (threadIdx.x >> 5);  // 0..4095
  int nt = wid & 255, ot = (wid >> 8) & 7, b = wid >> 11;
  int n0 = nt * 16, o0 = ot * 16;

  v8f acc{};
  for (int c0 = 0; c0 < kdim; c0 += 32) {
    v16bf a = load_a_bf(Wb, kdim, o0, c0, lane);
    v16bf bf = *(const v16bf*)(inB + ((size_t)b * NPT + n0 + r16) * ldin + c0 + hi * 16);
    acc = wmma_bf16(a, bf, acc);
  }
  int n = n0 + r16;
#pragma unroll
  for (int r = 0; r < 8; ++r) {
    int o = o0 + hi * 8 + r;
    float v = acc[r] + bias[o];
    if (bng) v = bng[o] * (v * BN_RS) + bnb[o];
    if (relu) v = fmaxf(v, 0.f);
    size_t oidx = ((size_t)b * NPT + n) * ldout + ocoff + o;
    if (outbf) outbf[oidx] = (__bf16)v;
    else       outf[oidx] = v;
  }
}

// ---------------------------------------------------------------------------
// Per-point: conf = sigmoid(zoom * 1.414*(Hs - Hs^2/N)), tiny MLP with BN/ReLU,
// LayerNorm, xr = x + n_feat -> fbf[b][n][0..127] (first half of concat).
// ---------------------------------------------------------------------------
__global__ void k_conf_mlp(const float* __restrict__ colsum, const float* __restrict__ zoom,
                           const float* __restrict__ x,
                           const float* nw1, const float* nb1, const float* ng1, const float* nbe1,
                           const float* nw2, const float* nb2, const float* ng2, const float* nbe2,
                           const float* nw3, const float* nb3, const float* ng3, const float* nbe3,
                           const float* nw4, const float* nb4,
                           const float* lng, const float* lnb,
                           __bf16* __restrict__ fbf) {
  int idx = blockIdx.x * blockDim.x + threadIdx.x;
  if (idx >= BB * NPT) return;
  int b = idx >> 12, n = idx & (NPT - 1);

  float c0[4];
#pragma unroll
  for (int h = 0; h < 4; ++h) {
    float Hs = colsum[((size_t)b * NH + h) * NPT + n];
    Hs = 1.414f * (Hs - Hs * Hs * (1.0f / (float)NPT));
    c0[h] = 1.0f / (1.0f + __expf(-zoom[h] * Hs));
  }
  float h1[32];
  for (int o = 0; o < 32; ++o) {
    float a = nb1[o];
#pragma unroll
    for (int i = 0; i < 4; ++i) a += nw1[o * 4 + i] * c0[i];
    h1[o] = fmaxf(ng1[o] * (a * BN_RS) + nbe1[o], 0.f);
  }
  float h2[64];
  for (int o = 0; o < 64; ++o) {
    float a = nb2[o];
    for (int i = 0; i < 32; ++i) a += nw2[o * 32 + i] * h1[i];
    h2[o] = fmaxf(ng2[o] * (a * BN_RS) + nbe2[o], 0.f);
  }
  float h3[128];
  for (int o = 0; o < 128; ++o) {
    float a = nb3[o];
    for (int i = 0; i < 64; ++i) a += nw3[o * 64 + i] * h2[i];
    h3[o] = fmaxf(ng3[o] * (a * BN_RS) + nbe3[o], 0.f);
  }
  float t4[128];
  float mu = 0.f;
  for (int o = 0; o < 128; ++o) {
    float a = nb4[o];
    for (int i = 0; i < 128; ++i) a += nw4[o * 128 + i] * h3[i];
    t4[o] = a;
    mu += a;
  }
  mu *= (1.0f / 128.0f);
  float var = 0.f;
  for (int o = 0; o < 128; ++o) { float dd = t4[o] - mu; var += dd * dd; }
  var *= (1.0f / 128.0f);
  float inv = rsqrtf(var + LN_EPS);
  for (int c = 0; c < 128; ++c) {
    float nf = (t4[c] - mu) * inv * lng[c] + lnb[c];
    float xr = x[((size_t)b * CCH + c) * NPT + n] + nf;
    fbf[((size_t)b * NPT + n) * 256 + c] = (__bf16)xr;
  }
}

// f2 [b][N][128] fp32 -> LayerNorm over c -> out [b][c][N]
__global__ void k_final_ln(const float* __restrict__ f2, const float* __restrict__ lng,
                           const float* __restrict__ lnb, float* __restrict__ out) {
  int idx = blockIdx.x * blockDim.x + threadIdx.x;
  if (idx >= BB * NPT) return;
  int b = idx >> 12, n = idx & (NPT - 1);
  const float* row = f2 + (size_t)idx * CCH;
  float mu = 0.f;
  for (int c = 0; c < CCH; ++c) mu += row[c];
  mu *= (1.0f / (float)CCH);
  float var = 0.f;
  for (int c = 0; c < CCH; ++c) { float d = row[c] - mu; var += d * d; }
  var *= (1.0f / (float)CCH);
  float inv = rsqrtf(var + LN_EPS);
  for (int c = 0; c < CCH; ++c)
    out[((size_t)b * CCH + c) * NPT + n] = (row[c] - mu) * inv * lng[c] + lnb[c];
}

// ---------------------------------------------------------------------------
extern "C" void kernel_launch(void* const* d_in, const int* in_sizes, int n_in,
                              void* d_out, int out_size, void* d_ws, size_t ws_size,
                              hipStream_t stream) {
  (void)in_sizes; (void)n_in; (void)out_size; (void)ws_size;
  const float* x    = (const float*)d_in[0];
  const float* wq   = (const float*)d_in[1];
  const float* bq   = (const float*)d_in[2];
  const float* wk   = (const float*)d_in[3];
  const float* bk   = (const float*)d_in[4];
  const float* wfc  = (const float*)d_in[5];
  const float* bfc  = (const float*)d_in[6];
  const float* zoom = (const float*)d_in[7];
  const float* nw1  = (const float*)d_in[8];
  const float* nb1  = (const float*)d_in[9];
  const float* ng1  = (const float*)d_in[10];
  const float* nbe1 = (const float*)d_in[11];
  const float* nw2  = (const float*)d_in[12];
  const float* nb2  = (const float*)d_in[13];
  const float* ng2  = (const float*)d_in[14];
  const float* nbe2 = (const float*)d_in[15];
  const float* nw3  = (const float*)d_in[16];
  const float* nb3  = (const float*)d_in[17];
  const float* ng3  = (const float*)d_in[18];
  const float* nbe3 = (const float*)d_in[19];
  const float* nw4  = (const float*)d_in[20];
  const float* nb4  = (const float*)d_in[21];
  const float* lng  = (const float*)d_in[22];
  const float* lnb  = (const float*)d_in[23];
  const float* cw1  = (const float*)d_in[24];
  const float* cb1  = (const float*)d_in[25];
  const float* cg1  = (const float*)d_in[26];
  const float* cbe1 = (const float*)d_in[27];
  const float* cw2  = (const float*)d_in[28];
  const float* cb2  = (const float*)d_in[29];
  float* out = (float*)d_out;

  char* p = (char*)d_ws;
  auto carve = [&](size_t bytes) -> char* {
    char* r = p;
    p += (bytes + 255) & ~(size_t)255;
    return r;
  };
  __bf16* xT     = (__bf16*)carve((size_t)BB * NPT * CCH * 2);        // [b][n][c]
  __bf16* Qbf    = (__bf16*)carve((size_t)BB * NH * NPT * HD * 2);    // [b][h][n][d]
  __bf16* Kbf    = (__bf16*)carve((size_t)BB * NH * NPT * HD * 2);    // [b][h][m][d]
  __bf16* Vt     = (__bf16*)carve((size_t)BB * NH * HD * NPT * 2);    // [b][h][d][m]
  __bf16* attbf  = (__bf16*)carve((size_t)BB * NPT * CCH * 2);        // [b][n][c]
  __bf16* fbf    = (__bf16*)carve((size_t)BB * NPT * 256 * 2);        // [b][n][256]
  __bf16* gbf    = (__bf16*)carve((size_t)BB * NPT * CCH * 2);        // [b][n][c]
  float*  f2     = (float*) carve((size_t)BB * NPT * CCH * 4);        // [b][n][c]
  float*  rowmax = (float*) carve((size_t)BB * NH * NPT * 4);
  float*  rowsum = (float*) carve((size_t)BB * NH * NPT * 4);
  float*  colsum = (float*) carve((size_t)BB * NH * NPT * 4);
  __bf16* wqb    = (__bf16*)carve((size_t)CCH * CCH * 2);
  __bf16* wkb    = (__bf16*)carve((size_t)CCH * CCH * 2);
  __bf16* wfcb   = (__bf16*)carve((size_t)CCH * CCH * 2);
  __bf16* cw1b   = (__bf16*)carve((size_t)CCH * 256 * 2);
  __bf16* cw2b   = (__bf16*)carve((size_t)CCH * CCH * 2);

  // 0. weight pre-conversion to bf16 (removes per-tile cvt storms)
  k_cvt_bf16<<<(CCH * CCH + 255) / 256, 256, 0, stream>>>(wq, wqb, CCH * CCH);
  k_cvt_bf16<<<(CCH * CCH + 255) / 256, 256, 0, stream>>>(wk, wkb, CCH * CCH);
  k_cvt_bf16<<<(CCH * CCH + 255) / 256, 256, 0, stream>>>(wfc, wfcb, CCH * CCH);
  k_cvt_bf16<<<(CCH * 256 + 255) / 256, 256, 0, stream>>>(cw1, cw1b, CCH * 256);
  k_cvt_bf16<<<(CCH * CCH + 255) / 256, 256, 0, stream>>>(cw2, cw2b, CCH * CCH);
  // 1. transpose x to bf16 [b][n][c]
  k_transpose_x<<<(BB * NPT * CCH) / 256, 256, 0, stream>>>(x, xT);
  // 2. Q/K projections (WMMA), scattered to per-head layout
  k_gemm_qk<<<1024, 128, 0, stream>>>(wqb, bq, xT, Qbf);
  k_gemm_qk<<<1024, 128, 0, stream>>>(wkb, bk, xT, Kbf);
  // 3. V transpose for PV B-fragments
  k_transpose_v<<<(BB * NH * NPT * HD) / 256, 256, 0, stream>>>(Kbf, Vt);
  // 4. attention pass 1 (async LDS staging of K)
  k_attn_lse<<<512, 128, 0, stream>>>(Qbf, Kbf, rowmax, rowsum);
  // 5. attention pass 2: P@V + column sums of prob
  k_zero<<<(BB * NH * NPT + 255) / 256, 256, 0, stream>>>(colsum, BB * NH * NPT);
  k_attn_pv<<<512, 128, 0, stream>>>(Qbf, Kbf, Vt, rowmax, rowsum, colsum, attbf);
  // 6. c_feat = wfc@att + bfc -> upper half of concat buffer
  k_gemm_row<<<1024, 128, 0, stream>>>(wfcb, bfc, nullptr, nullptr, attbf, CCH, CCH,
                                       fbf, nullptr, 256, 128, 0);
  // 7. conf -> MLP -> LN -> xr = x + n_feat -> lower half of concat buffer
  k_conf_mlp<<<(BB * NPT + 127) / 128, 128, 0, stream>>>(
      colsum, zoom, x, nw1, nb1, ng1, nbe1, nw2, nb2, ng2, nbe2,
      nw3, nb3, ng3, nbe3, nw4, nb4, lng, lnb, fbf);
  // 8. g = relu(bn(cw1@f + cb1))
  k_gemm_row<<<1024, 128, 0, stream>>>(cw1b, cb1, cg1, cbe1, fbf, 256, 256,
                                       gbf, nullptr, CCH, 0, 1);
  // 9. f2 = cw2@g + cb2 (fp32)
  k_gemm_row<<<1024, 128, 0, stream>>>(cw2b, cb2, nullptr, nullptr, gbf, CCH, CCH,
                                       nullptr, f2, CCH, 0, 0);
  // 10. final LayerNorm -> out [b][c][n]
  k_final_ln<<<(BB * NPT + 127) / 128, 128, 0, stream>>>(f2, lng, lnb, out);
}
